// RandomGraphVariant_43276090474723
// MI455X (gfx1250) — compile-verified, matrix operand order checked by reading, not run on hardware
//
#include <hip/hip_runtime.h>
#include <math.h>

typedef __attribute__((ext_vector_type(16))) _Float16 v16h;
typedef __attribute__((ext_vector_type(8)))  float    v8f;

#define B_    512
#define NPG_  34
#define NT_   (B_*NPG_)
#define NLOC_ 33
#define FEAT_ 1152
#define GPD_  2048
#define HID_  1024
#define HAD_  512          // H*AD = 8*64
#define LLM_  4096
#define K1_   24576        // 32*C
#define EPG_  272          // edges per graph
#define ETOT_ (B_*EPG_)

// ---------------------------------------------------------------------------
// f32 [K,N] -> f16 [N,K] transpose + convert (LDS-tiled, both sides coalesced)
// ---------------------------------------------------------------------------
__global__ __launch_bounds__(256)
void transpose_cvt(const float* __restrict__ W, _Float16* __restrict__ Wt,
                   int K, int N) {
    __shared__ _Float16 tile[64][66];                 // odd-ish stride -> no bank conflicts
    const int n0 = blockIdx.x * 64;
    const int k0 = blockIdx.y * 64;
    const int t  = threadIdx.x;
    const int j  = t & 63;          // fast index
    const int r4 = t >> 6;          // 0..3
    for (int i = r4; i < 64; i += 4)
        tile[i][j] = (_Float16)W[(size_t)(k0 + i) * N + n0 + j];
    __syncthreads();
    for (int jj = r4; jj < 64; jj += 4)
        Wt[(size_t)(n0 + jj) * K + k0 + j] = tile[j][jj];
}

// ---------------------------------------------------------------------------
// AdaptiveAvgPool3d (8,8,4)->(4,4,2) == exact 2x2x2 mean, f32 -> f16
// out[b, c*32 + oi*8 + oj*2 + ok]
// ---------------------------------------------------------------------------
__global__ __launch_bounds__(256)
void pool_cvt(const float* __restrict__ g, _Float16* __restrict__ pooled) {
    const int o = blockIdx.x * 256 + threadIdx.x;     // < 512*24576, exact grid
    const int b = o / K1_;
    const int r = o - b * K1_;
    const int c = r >> 5;
    const int p = r & 31;
    const int oi = p >> 3, oj = (p >> 1) & 3, ok = p & 1;
    const float* base = g + (size_t)b * 196608 + c * 256 + oi * 64 + oj * 8 + ok * 2;
    float s = 0.f;
#pragma unroll
    for (int di = 0; di < 2; ++di)
#pragma unroll
        for (int dj = 0; dj < 2; ++dj) {
            float2 v = *(const float2*)(base + di * 32 + dj * 4);
            s += v.x + v.y;
        }
    pooled[o] = (_Float16)(s * 0.125f);
}

// ---------------------------------------------------------------------------
// local_features f32 [B,33,1152] -> f16 node rows (34b+1+i)
// ---------------------------------------------------------------------------
__global__ __launch_bounds__(256)
void local_cvt(const float* __restrict__ lf, _Float16* __restrict__ nodes) {
    const int i = blockIdx.x * 256 + threadIdx.x;     // exact: 512*33*1152
    const int b = i / (NLOC_ * FEAT_);
    const int r = i - b * (NLOC_ * FEAT_);
    const int n = r / FEAT_;
    const int f = r - n * FEAT_;
    nodes[((size_t)b * NPG_ + 1 + n) * FEAT_ + f] = (_Float16)lf[i];
}

// ---------------------------------------------------------------------------
// WMMA f16 GEMM:  C[M,N] = A[M,K](f16,row) * Bt[N,K](f16,row) (+bias)(+relu)
// 256 threads = 8 waves (2m x 4n). BK=32 (one v_wmma_f32_16x16x32_f16 per
// fragment pair). Register-staged prefetch of next K tile behind the WMMAs.
// OUT_F32 selects the single (compile-time) output path; ostride is the
// element stride between consecutive logical C rows (supports the scattered
// node-row write of GEMM2).
// ---------------------------------------------------------------------------
template<int BM, int BN, bool RELU, bool HAS_BIAS, bool OUT_F32>
__global__ __launch_bounds__(256)
void gemm_wmma(const _Float16* __restrict__ A, const _Float16* __restrict__ Bt,
               const float* __restrict__ bias, void* __restrict__ outP,
               long long ostride, int M, int N, int K) {
    constexpr int BK  = 32;
    constexpr int LDT = BK + 16;          // 48 halves = 96B rows, 32B aligned
    constexpr int WM = 2, WN = 4;
    constexpr int WTM = BM / WM;          // 64 / 32
    constexpr int WTN = BN / WN;          // 32 / 16
    constexpr int MT = WTM / 16;          // 4 / 2
    constexpr int NTL = WTN / 16;         // 2 / 1
    constexpr int CHA = (BM * 4) / 256;   // 16B chunks per thread for A tile
    constexpr int CHB = (BN * 4) / 256;

    __shared__ _Float16 ldsA[BM * LDT];
    __shared__ _Float16 ldsB[BN * LDT];

    const int t    = threadIdx.x;
    const int lane = t & 31;
    const int wave = t >> 5;
    const int wm = wave >> 2, wn = wave & 3;
    const int mBase = blockIdx.x * BM;
    const int nBase = blockIdx.y * BN;
    const int lrow = lane & 15;
    const int kh   = (lane >> 4) << 4;    // K half offset: 0 or 16

    v8f zero = {};
    v8f acc[MT][NTL];
#pragma unroll
    for (int i = 0; i < MT; ++i)
#pragma unroll
        for (int j = 0; j < NTL; ++j) acc[i][j] = zero;

    const int ktiles = K / BK;

    // prologue: tile 0 -> LDS
#pragma unroll
    for (int u = 0; u < CHA; ++u) {
        int c = t + u * 256, r = c >> 2, q = c & 3;
        *(float4*)&ldsA[r * LDT + q * 8] =
            *(const float4*)&A[(size_t)(mBase + r) * K + q * 8];
    }
#pragma unroll
    for (int u = 0; u < CHB; ++u) {
        int c = t + u * 256, r = c >> 2, q = c & 3;
        *(float4*)&ldsB[r * LDT + q * 8] =
            *(const float4*)&Bt[(size_t)(nBase + r) * K + q * 8];
    }
    __syncthreads();

    for (int kt = 0; kt < ktiles; ++kt) {
        // stage next tile into registers (latency hidden behind WMMAs)
        float4 sa[CHA], sb[CHB];
        const bool more = (kt + 1) < ktiles;
        if (more) {
            const int k0 = (kt + 1) * BK;
#pragma unroll
            for (int u = 0; u < CHA; ++u) {
                int c = t + u * 256, r = c >> 2, q = c & 3;
                sa[u] = *(const float4*)&A[(size_t)(mBase + r) * K + k0 + q * 8];
            }
#pragma unroll
            for (int u = 0; u < CHB; ++u) {
                int c = t + u * 256, r = c >> 2, q = c & 3;
                sb[u] = *(const float4*)&Bt[(size_t)(nBase + r) * K + k0 + q * 8];
            }
        }

        // fragments from LDS (contiguous k order for A and B: row=lane&15, k = 16*(lane>>4) + s)
        v16h af[MT], bf[NTL];
#pragma unroll
        for (int i = 0; i < MT; ++i)
            af[i] = *(const v16h*)&ldsA[(wm * WTM + i * 16 + lrow) * LDT + kh];
#pragma unroll
        for (int j = 0; j < NTL; ++j)
            bf[j] = *(const v16h*)&ldsB[(wn * WTN + j * 16 + lrow) * LDT + kh];

#pragma unroll
        for (int i = 0; i < MT; ++i)
#pragma unroll
            for (int j = 0; j < NTL; ++j)
                acc[i][j] = __builtin_amdgcn_wmma_f32_16x16x32_f16(
                    false, af[i], false, bf[j], (short)0, acc[i][j], false, false);

        __syncthreads();
        if (more) {
#pragma unroll
            for (int u = 0; u < CHA; ++u) {
                int c = t + u * 256, r = c >> 2, q = c & 3;
                *(float4*)&ldsA[r * LDT + q * 8] = sa[u];
            }
#pragma unroll
            for (int u = 0; u < CHB; ++u) {
                int c = t + u * 256, r = c >> 2, q = c & 3;
                *(float4*)&ldsB[r * LDT + q * 8] = sb[u];
            }
        }
        __syncthreads();
    }

    // epilogue: C/D layout -> row = 8*(lane>>4)+v, col = lane&15.
    // Single compile-time output path; base address hoisted, pointer stepping
    // (no per-element 64-bit multiplies, no branches).
    const int orow0 = mBase + wm * WTM + (lane >> 4) * 8;
    const int ocol0 = nBase + wn * WTN;
#pragma unroll
    for (int i = 0; i < MT; ++i) {
#pragma unroll
        for (int j = 0; j < NTL; ++j) {
            const int col = ocol0 + j * 16 + lrow;
            const float bv = HAS_BIAS ? bias[col] : 0.f;
            const size_t base = (size_t)(orow0 + i * 16) * (size_t)ostride + col;
            if (OUT_F32) {
                float* p = (float*)outP + base;
#pragma unroll
                for (int v = 0; v < 8; ++v) {
                    float x = acc[i][j][v] + bv;
                    if (RELU) x = fmaxf(x, 0.f);
                    *p = x;
                    p += ostride;
                }
            } else {
                _Float16* p = (_Float16*)outP + base;
#pragma unroll
                for (int v = 0; v < 8; ++v) {
                    float x = acc[i][j][v] + bv;
                    if (RELU) x = fmaxf(x, 0.f);
                    *p = (_Float16)x;
                    p += ostride;
                }
            }
        }
    }
}

// ---------------------------------------------------------------------------
// GAT attention: one workgroup per graph (34 nodes, 272 edges, 8 heads x 64).
// Everything in LDS; deterministic (no atomics). Output agg as f16 for GEMM5.
// ---------------------------------------------------------------------------
__global__ __launch_bounds__(512)
void gat_attention(const float* __restrict__ z, const int* __restrict__ ei,
                   const float* __restrict__ a_src, const float* __restrict__ a_dst,
                   _Float16* __restrict__ aggH) {
    __shared__ float zs[NPG_ * HAD_];     // 69632 B
    __shared__ float aggs[NPG_ * HAD_];   // 69632 B
    __shared__ float es[EPG_ * 8];        // 8704 B
    __shared__ float ssrc[NPG_ * 8];
    __shared__ float sdst[NPG_ * 8];
    __shared__ int   sl[EPG_];
    __shared__ int   dl[EPG_];
    __shared__ float aS[HAD_];
    __shared__ float aD[HAD_];

    const int b = blockIdx.x;
    const int t = threadIdx.x;

    for (int i = t; i < NPG_ * HAD_; i += 512) {
        zs[i]   = z[(size_t)b * NPG_ * HAD_ + i];
        aggs[i] = 0.f;
    }
    aS[t] = a_src[t];
    aD[t] = a_dst[t];
    for (int e = t; e < EPG_; e += 512) {
        sl[e] = ei[b * EPG_ + e]          - b * NPG_;
        dl[e] = ei[ETOT_ + b * EPG_ + e]  - b * NPG_;
    }
    __syncthreads();

    // per-node attention pre-scores: s_src[n,h] = z[n,h,:].a_src[h,:]
    if (t < NPG_ * 8) {
        const int n = t >> 3, h = t & 7;
        float s1 = 0.f, s2 = 0.f;
        const float* zr = &zs[n * HAD_ + h * 64];
#pragma unroll 8
        for (int d = 0; d < 64; ++d) {
            s1 += zr[d] * aS[h * 64 + d];
            s2 += zr[d] * aD[h * 64 + d];
        }
        ssrc[t] = s1;
        sdst[t] = s2;
    }
    __syncthreads();

    // edge scores with leaky-relu(0.2)
    for (int i = t; i < EPG_ * 8; i += 512) {
        const int e = i >> 3, h = i & 7;
        const float v = ssrc[sl[e] * 8 + h] + sdst[dl[e] * 8 + h];
        es[i] = v > 0.f ? v : 0.2f * v;
    }
    __syncthreads();

    // neighbor softmax per (dst node, head): one thread per segment
    if (t < NPG_ * 8) {
        const int n = t >> 3, h = t & 7;
        float m = -1e30f;
        for (int e = 0; e < EPG_; ++e)
            if (dl[e] == n) m = fmaxf(m, es[e * 8 + h]);
        float den = 0.f;
        for (int e = 0; e < EPG_; ++e)
            if (dl[e] == n) den += expf(es[e * 8 + h] - m);
        const float inv = 1.f / (den + 1e-9f);
        for (int e = 0; e < EPG_; ++e)
            if (dl[e] == n) es[e * 8 + h] = expf(es[e * 8 + h] - m) * inv;
    }
    __syncthreads();

    // aggregate: thread t owns channel t (head = t>>6); race-free
    {
        const int h = t >> 6;
        for (int e = 0; e < EPG_; ++e)
            aggs[dl[e] * HAD_ + t] += es[e * 8 + h] * zs[sl[e] * HAD_ + t];
    }
    __syncthreads();

    for (int i = t; i < NPG_ * HAD_; i += 512)
        aggH[(size_t)b * NPG_ * HAD_ + i] = (_Float16)aggs[i];
}

// ---------------------------------------------------------------------------
extern "C" void kernel_launch(void* const* d_in, const int* in_sizes, int n_in,
                              void* d_out, int out_size, void* d_ws, size_t ws_size,
                              hipStream_t stream) {
    (void)in_sizes; (void)n_in; (void)out_size; (void)ws_size;
    const float* lf    = (const float*)d_in[0];
    const float* gf    = (const float*)d_in[1];
    const int*   ei    = (const int*)d_in[2];
    const float* W1g   = (const float*)d_in[3];
    const float* b1g   = (const float*)d_in[4];
    const float* W2g   = (const float*)d_in[5];
    const float* b2g   = (const float*)d_in[6];
    const float* Wp    = (const float*)d_in[7];
    const float* bp    = (const float*)d_in[8];
    const float* Wg    = (const float*)d_in[9];
    const float* a_src = (const float*)d_in[10];
    const float* a_dst = (const float*)d_in[11];
    const float* Wllm  = (const float*)d_in[12];
    const float* bllm  = (const float*)d_in[13];
    float* out = (float*)d_out;

    char* ws = (char*)d_ws;
    size_t off = 0;
    auto alloc = [&](size_t bytes) -> void* {
        void* p = ws + off;
        off = (off + bytes + 255) & ~(size_t)255;
        return p;
    };
    _Float16* W1gT   = (_Float16*)alloc((size_t)K1_ * GPD_ * 2);     // 100.7 MB
    _Float16* W2gT   = (_Float16*)alloc((size_t)GPD_ * FEAT_ * 2);
    _Float16* WpT    = (_Float16*)alloc((size_t)FEAT_ * HID_ * 2);
    _Float16* WgT    = (_Float16*)alloc((size_t)HID_ * HAD_ * 2);
    _Float16* WllmT  = (_Float16*)alloc((size_t)HAD_ * LLM_ * 2);
    _Float16* pooled = (_Float16*)alloc((size_t)B_ * K1_ * 2);
    _Float16* y1     = (_Float16*)alloc((size_t)B_ * GPD_ * 2);
    _Float16* nodes  = (_Float16*)alloc((size_t)NT_ * FEAT_ * 2);
    _Float16* hbuf   = (_Float16*)alloc((size_t)NT_ * HID_ * 2);
    float*    zbuf   = (float*)   alloc((size_t)NT_ * HAD_ * 4);
    _Float16* aggH   = (_Float16*)alloc((size_t)NT_ * HAD_ * 2);

    // weight transpose + f32->f16 conversion (W1g_f16 = 100MB -> resident in 192MB L2)
    transpose_cvt<<<dim3(GPD_/64,  K1_/64),  256, 0, stream>>>(W1g,  W1gT,  K1_,  GPD_);
    transpose_cvt<<<dim3(FEAT_/64, GPD_/64), 256, 0, stream>>>(W2g,  W2gT,  GPD_, FEAT_);
    transpose_cvt<<<dim3(HID_/64,  FEAT_/64),256, 0, stream>>>(Wp,   WpT,   FEAT_, HID_);
    transpose_cvt<<<dim3(HAD_/64,  HID_/64), 256, 0, stream>>>(Wg,   WgT,   HID_,  HAD_);
    transpose_cvt<<<dim3(LLM_/64,  HAD_/64), 256, 0, stream>>>(Wllm, WllmT, HAD_,  LLM_);

    // pooling + node assembly
    pool_cvt <<<(B_ * K1_) / 256,           256, 0, stream>>>(gf, pooled);
    local_cvt<<<(B_ * NLOC_ * FEAT_) / 256, 256, 0, stream>>>(lf, nodes);

    // GEMM1: y1 = relu(pooled @ W1g + b1g)          [512, 2048] f16
    gemm_wmma<64, 64, true, true, false><<<dim3(B_/64, GPD_/64), 256, 0, stream>>>(
        pooled, W1gT, b1g, y1, GPD_, B_, GPD_, K1_);
    // GEMM2: out_global -> node rows 34*b            [512, 1152] f16 scattered
    gemm_wmma<64, 64, false, true, false><<<dim3(B_/64, FEAT_/64), 256, 0, stream>>>(
        y1, W2gT, b2g, nodes, (long long)NPG_ * FEAT_, B_, FEAT_, GPD_);
    // GEMM3: h = relu(nodes @ Wp + bp)               [17408, 1024] f16
    gemm_wmma<128, 128, true, true, false><<<dim3(NT_/128, HID_/128), 256, 0, stream>>>(
        nodes, WpT, bp, hbuf, HID_, NT_, HID_, FEAT_);
    // GEMM4: z = h @ Wg (f32 out for attention)      [17408, 512] f32
    gemm_wmma<128, 128, false, false, true><<<dim3(NT_/128, HAD_/128), 256, 0, stream>>>(
        hbuf, WgT, nullptr, zbuf, HAD_, NT_, HAD_, HID_);
    // GAT: neighbor softmax + aggregation, one WG per graph
    gat_attention<<<B_, 512, 0, stream>>>(zbuf, ei, a_src, a_dst, aggH);
    // GEMM5: out = agg @ Wllm + bllm                 [17408, 4096] f32
    gemm_wmma<128, 128, false, true, true><<<dim3(NT_/128, LLM_/128), 256, 0, stream>>>(
        aggH, WllmT, bllm, out, LLM_, NT_, LLM_, HAD_);
}
